// GCN3_6158983102957
// MI455X (gfx1250) — matrix operand order repaired and checked by reading, not compiled, not run on hardware
//
#include <hip/hip_runtime.h>
#include <hip/hip_bf16.h>
#include <stdint.h>

// ---------------------------------------------------------------------------
// Types for CDNA5 WMMA
// ---------------------------------------------------------------------------
typedef __attribute__((ext_vector_type(16))) __bf16 v16bf;
typedef __attribute__((ext_vector_type(8)))  float  v8f;

union Frag {            // one WMMA operand: 16 bf16 per lane = 32 B = 2 x b128
    uint4 q[2];
    v16bf v;
};

// ---------------------------------------------------------------------------
// fp32 -> bf16 conversion
// ---------------------------------------------------------------------------
__global__ void k_cvt_bf16(const float* __restrict__ in, __bf16* __restrict__ out,
                           size_t n) {
    size_t i = (size_t)blockIdx.x * blockDim.x + threadIdx.x;
    if (i < n) out[i] = (__bf16)in[i];
}

// ---------------------------------------------------------------------------
// degree (with self loop) -> dinv = rsqrt(1 + deg)
// ---------------------------------------------------------------------------
__global__ void k_deg(const long long* __restrict__ dst, float* __restrict__ deg,
                      int E) {
    int i = blockIdx.x * blockDim.x + threadIdx.x;
    if (i < E) atomicAdd(&deg[(int)dst[i]], 1.0f);
}

__global__ void k_dinv(float* __restrict__ deg, int n) {
    int i = blockIdx.x * blockDim.x + threadIdx.x;
    if (i < n) deg[i] = rsqrtf(1.0f + deg[i]);   // in-place: deg becomes dinv
}

// ---------------------------------------------------------------------------
// BF16 WMMA GEMM: C[Mpad,N] = A[Mpad,K] * B[K,N]
//   block = 256 threads (8 wave32), block tile = 128 rows x 64 cols
//   wave tile = 32x32 (2 A-frags x 2 B-frags -> 4 v_wmma per 32-K step)
//   LDS holds tiles pre-swizzled into per-lane fragment slots (12-dword pitch)
//   requires: Mpad % 128 == 0, N % 64 == 0, K % 32 == 0
// ---------------------------------------------------------------------------
__global__ void __launch_bounds__(256)
k_gemm_bf16_wmma(const __bf16* __restrict__ A, const __bf16* __restrict__ B,
                 float* __restrict__ C, int N, int K) {
    // fragment-order staging buffers: [tile][lane][8 dwords + 4 pad]
    __shared__ uint32_t A_sw[8][32][12];   // 8 row-tiles of 16 rows   (12 KB)
    __shared__ uint32_t B_sw[4][32][12];   // 4 col-tiles of 16 cols   ( 6 KB)

    const int tid  = threadIdx.x;
    const int lane = tid & 31;
    const int wave = tid >> 5;            // 0..7
    const int wr   = wave >> 1;           // 0..3 : 32-row strip
    const int wc   = wave & 1;            // 0..1 : 32-col strip
    const int row0 = blockIdx.x * 128;
    const int colB = blockIdx.y * 64;
    const int half = lane >> 4;

    // ---- staging index math (loop-invariant) ------------------------------
    // A: thread stages one b128 pair: row ra (0..127), kcol group kcb in {0,4,8,12}x2
    const int ra  = tid >> 1;                       // 0..127
    const int art = ra >> 4;                        // row-tile 0..7
    const int aar = ra & 15;                        // row in tile
    // B: thread stages K-pair k0=2*kp over 4 columns 4*cp..4*cp+3
    const int kp  = tid >> 4;                       // 0..15
    const int cp  = tid & 15;                       // 0..15
    const int bhalf = (kp >> 2) & 1;
    const int bv    = (kp & 3) + ((kp >> 3) << 2);
    const int bct   = cp >> 2;                      // col-tile 0..3
    const int bln0  = bhalf * 16 + ((4 * (cp & 3)) & 15);

    const uint32_t* Ag = (const uint32_t*)A;
    const uint32_t* Bg = (const uint32_t*)B;

    v8f acc00 = {}, acc01 = {}, acc10 = {}, acc11 = {};

    for (int k0 = 0; k0 < K; k0 += 32) {
        // ---- stage A: 128x32 bf16 = 2048 dwords, 2 x b128 per thread ------
        {
            size_t rowbase = ((size_t)(row0 + ra) * K + k0) >> 1;   // dword idx
            #pragma unroll
            for (int g = 0; g < 2; ++g) {
                int kcb = 8 * (tid & 1) + 4 * g;                    // 0/4/8/12
                uint4 v = *(const uint4*)(Ag + rowbase + kcb);
                int ahalf = (kcb >> 2) & 1;
                int jb    = (kcb >> 3) * 4;                         // 0 or 4
                *(uint4*)&A_sw[art][ahalf * 16 + aar][jb] = v;
            }
        }
        // ---- stage B: 32x64 bf16 = 1024 dwords, repack K-major via v_perm -
        {
            int krow = k0 + 2 * kp;
            size_t b0 = ((size_t)krow * N + colB) >> 1;             // dword idx
            size_t b1 = b0 + (size_t)(N >> 1);
            uint2 d0 = *(const uint2*)(Bg + b0 + 2 * cp);           // row k,   cols 4cp..4cp+3
            uint2 d1 = *(const uint2*)(Bg + b1 + 2 * cp);           // row k+1
            uint32_t p0 = __builtin_amdgcn_perm(d1.x, d0.x, 0x05040100u); // col 4cp
            uint32_t p1 = __builtin_amdgcn_perm(d1.x, d0.x, 0x07060302u); // col 4cp+1
            uint32_t p2 = __builtin_amdgcn_perm(d1.y, d0.y, 0x05040100u); // col 4cp+2
            uint32_t p3 = __builtin_amdgcn_perm(d1.y, d0.y, 0x07060302u); // col 4cp+3
            B_sw[bct][bln0 + 0][bv] = p0;
            B_sw[bct][bln0 + 1][bv] = p1;
            B_sw[bct][bln0 + 2][bv] = p2;
            B_sw[bct][bln0 + 3][bv] = p3;
        }
        __syncthreads();

        // ---- fragment loads: 2 x ds_load_b128 each ------------------------
        Frag a0, a1, bf0, bf1;
        {
            const uint32_t* s;
            s = A_sw[wr * 2 + 0][lane]; a0.q[0] = *(const uint4*)&s[0]; a0.q[1] = *(const uint4*)&s[4];
            s = A_sw[wr * 2 + 1][lane]; a1.q[0] = *(const uint4*)&s[0]; a1.q[1] = *(const uint4*)&s[4];
            s = B_sw[wc * 2 + 0][lane]; bf0.q[0] = *(const uint4*)&s[0]; bf0.q[1] = *(const uint4*)&s[4];
            s = B_sw[wc * 2 + 1][lane]; bf1.q[0] = *(const uint4*)&s[0]; bf1.q[1] = *(const uint4*)&s[4];
        }

        acc00 = __builtin_amdgcn_wmma_f32_16x16x32_bf16(false, a0.v, false, bf0.v, (short)0, acc00, false, false);
        acc01 = __builtin_amdgcn_wmma_f32_16x16x32_bf16(false, a0.v, false, bf1.v, (short)0, acc01, false, false);
        acc10 = __builtin_amdgcn_wmma_f32_16x16x32_bf16(false, a1.v, false, bf0.v, (short)0, acc10, false, false);
        acc11 = __builtin_amdgcn_wmma_f32_16x16x32_bf16(false, a1.v, false, bf1.v, (short)0, acc11, false, false);
        __syncthreads();
    }

    // ---- store D tiles: VGPR i -> row = half*8 + i, col = lane&15 ---------
    const int rW = row0 + wr * 32;
    const int cW = colB + wc * 32 + (lane & 15);
    #pragma unroll
    for (int i = 0; i < 8; ++i) {
        int rr = half * 8 + i;
        C[(size_t)(rW + rr)      * N + cW]      = acc00[i];
        C[(size_t)(rW + rr)      * N + cW + 16] = acc01[i];
        C[(size_t)(rW + 16 + rr) * N + cW]      = acc10[i];
        C[(size_t)(rW + 16 + rr) * N + cW + 16] = acc11[i];
    }
}

// ---------------------------------------------------------------------------
// edge aggregation: agg[dst] += h[src] * dinv[src]*dinv[dst]   (wave per edge)
// ---------------------------------------------------------------------------
__global__ void k_edge_agg(const float* __restrict__ h, float* __restrict__ agg,
                           const float* __restrict__ dinv,
                           const long long* __restrict__ src,
                           const long long* __restrict__ dst, int E, int F) {
    int warp = blockIdx.x * (blockDim.x >> 5) + (threadIdx.x >> 5);
    int lane = threadIdx.x & 31;
    if (warp >= E) return;
    int s = (int)src[warp];
    int d = (int)dst[warp];
    float nrm = dinv[s] * dinv[d];
    const float4* hs = (const float4*)(h + (size_t)s * F);
    float* ad = agg + (size_t)d * F;
    for (int f = lane * 4; f < F; f += 128) {
        float4 v = hs[f >> 2];
        atomicAdd(&ad[f + 0], v.x * nrm);
        atomicAdd(&ad[f + 1], v.y * nrm);
        atomicAdd(&ad[f + 2], v.z * nrm);
        atomicAdd(&ad[f + 3], v.w * nrm);
    }
}

// ---------------------------------------------------------------------------
// post: out_bf16 = relu(agg + h*dinv^2 + b)   (feeds next layer's GEMM)
// ---------------------------------------------------------------------------
__global__ void k_post_relu(const float* __restrict__ h, const float* __restrict__ agg,
                            const float* __restrict__ dinv, const float* __restrict__ bias,
                            __bf16* __restrict__ outb, int M, int F) {
    size_t i = (size_t)blockIdx.x * blockDim.x + threadIdx.x;
    size_t total = (size_t)M * F;
    if (i >= total) return;
    int row = (int)(i / F);
    int col = (int)(i % F);
    float di = dinv[row];
    float v = agg[i] + h[i] * di * di + bias[col];
    outb[i] = (__bf16)fmaxf(v, 0.0f);
}

// ---------------------------------------------------------------------------
// final: z = agg + h*dinv^2 + b ; p = softmax(z) ; out = log_softmax(p)
//   F == 64, wave per row, 2 cols per lane, wave32 shuffles
// ---------------------------------------------------------------------------
__global__ void k_post_softmax(const float* __restrict__ h, const float* __restrict__ agg,
                               const float* __restrict__ dinv, const float* __restrict__ bias,
                               float* __restrict__ out, int M) {
    int row  = blockIdx.x * (blockDim.x >> 5) + (threadIdx.x >> 5);
    int lane = threadIdx.x & 31;
    if (row >= M) return;
    float di = dinv[row];
    float dd = di * di;
    size_t base = (size_t)row * 64;

    float z0 = agg[base + lane]      + h[base + lane]      * dd + bias[lane];
    float z1 = agg[base + 32 + lane] + h[base + 32 + lane] * dd + bias[32 + lane];

    float m = fmaxf(z0, z1);
    #pragma unroll
    for (int o = 16; o > 0; o >>= 1) m = fmaxf(m, __shfl_xor(m, o, 32));
    float e0 = __expf(z0 - m), e1 = __expf(z1 - m);
    float s = e0 + e1;
    #pragma unroll
    for (int o = 16; o > 0; o >>= 1) s += __shfl_xor(s, o, 32);
    float p0 = e0 / s, p1 = e1 / s;

    // log_softmax applied to p (faithful to reference)
    float m2 = fmaxf(p0, p1);
    #pragma unroll
    for (int o = 16; o > 0; o >>= 1) m2 = fmaxf(m2, __shfl_xor(m2, o, 32));
    float q0 = __expf(p0 - m2), q1 = __expf(p1 - m2);
    float s2 = q0 + q1;
    #pragma unroll
    for (int o = 16; o > 0; o >>= 1) s2 += __shfl_xor(s2, o, 32);
    float l = m2 + __logf(s2);

    out[base + lane]      = p0 - l;
    out[base + 32 + lane] = p1 - l;
}

// ---------------------------------------------------------------------------
// host-side orchestration
// ---------------------------------------------------------------------------
static inline size_t alignup(size_t x) { return (x + 255) & ~(size_t)255; }

extern "C" void kernel_launch(void* const* d_in, const int* in_sizes, int n_in,
                              void* d_out, int out_size, void* d_ws, size_t ws_size,
                              hipStream_t stream) {
    const float*     x  = (const float*)d_in[0];       // [M,512]
    const long long* ei = (const long long*)d_in[1];   // [2,E] int64
    const float*     W1 = (const float*)d_in[2];       // [512,256]
    const float*     b1 = (const float*)d_in[3];
    const float*     W2 = (const float*)d_in[4];       // [256,256]
    const float*     b2 = (const float*)d_in[5];
    const float*     W3 = (const float*)d_in[6];       // [256,64]
    const float*     b3 = (const float*)d_in[7];
    float* out = (float*)d_out;

    const int M    = in_sizes[0] / 512;                // 50000
    const int Mpad = (M + 127) & ~127;                 // 50048 (GEMM row padding)
    const int E    = in_sizes[1] / 2;                  // 800000
    const long long* src = ei;
    const long long* dst = ei + E;

    // ---- workspace carve-out (buffers reused across layers) --------------
    char* ws = (char*)d_ws;
    size_t off = 0;
    float*  dinv = (float*)(ws + off);  off = alignup(off + (size_t)M * 4);
    __bf16* w1b  = (__bf16*)(ws + off); off = alignup(off + (size_t)512 * 256 * 2);
    __bf16* w2b  = (__bf16*)(ws + off); off = alignup(off + (size_t)256 * 256 * 2);
    __bf16* w3b  = (__bf16*)(ws + off); off = alignup(off + (size_t)256 * 64 * 2);
    __bf16* xb   = (__bf16*)(ws + off); off = alignup(off + (size_t)Mpad * 512 * 2); // bf16 acts (all layers)
    float*  hbuf = (float*)(ws + off);  off = alignup(off + (size_t)Mpad * 256 * 4);
    float*  abuf = (float*)(ws + off);  off = alignup(off + (size_t)M * 256 * 4);
    (void)ws_size; (void)n_in; (void)out_size;

    // ---- degree / dinv ----------------------------------------------------
    hipMemsetAsync(dinv, 0, (size_t)M * 4, stream);
    k_deg<<<dim3((E + 255) / 256), dim3(256), 0, stream>>>(dst, dinv, E);
    k_dinv<<<dim3((M + 255) / 256), dim3(256), 0, stream>>>(dinv, M);

    // ---- bf16 conversions -------------------------------------------------
    {
        size_t n = (size_t)M * 512;
        k_cvt_bf16<<<dim3((unsigned)((n + 255) / 256)), dim3(256), 0, stream>>>(x, xb, n);
        k_cvt_bf16<<<dim3((512 * 256 + 255) / 256), dim3(256), 0, stream>>>(W1, w1b, 512 * 256);
        k_cvt_bf16<<<dim3((256 * 256 + 255) / 256), dim3(256), 0, stream>>>(W2, w2b, 256 * 256);
        k_cvt_bf16<<<dim3((256 * 64 + 255) / 256), dim3(256), 0, stream>>>(W3, w3b, 256 * 64);
    }

    const int EDGE_BLOCKS = (E + 7) / 8;   // 8 waves (edges) per 256-thread block

    // ---- layer 1: 512 -> 256 ---------------------------------------------
    k_gemm_bf16_wmma<<<dim3(Mpad / 128, 256 / 64), dim3(256), 0, stream>>>(
        xb, w1b, hbuf, 256, 512);
    hipMemsetAsync(abuf, 0, (size_t)M * 256 * 4, stream);
    k_edge_agg<<<dim3(EDGE_BLOCKS), dim3(256), 0, stream>>>(hbuf, abuf, dinv, src, dst, E, 256);
    {
        size_t n = (size_t)M * 256;
        k_post_relu<<<dim3((unsigned)((n + 255) / 256)), dim3(256), 0, stream>>>(
            hbuf, abuf, dinv, b1, xb, M, 256);   // xb reused: bf16 activations
    }

    // ---- layer 2: 256 -> 256 ---------------------------------------------
    k_gemm_bf16_wmma<<<dim3(Mpad / 128, 256 / 64), dim3(256), 0, stream>>>(
        xb, w2b, hbuf, 256, 256);
    hipMemsetAsync(abuf, 0, (size_t)M * 256 * 4, stream);
    k_edge_agg<<<dim3(EDGE_BLOCKS), dim3(256), 0, stream>>>(hbuf, abuf, dinv, src, dst, E, 256);
    {
        size_t n = (size_t)M * 256;
        k_post_relu<<<dim3((unsigned)((n + 255) / 256)), dim3(256), 0, stream>>>(
            hbuf, abuf, dinv, b2, xb, M, 256);
    }

    // ---- layer 3: 256 -> 64 ----------------------------------------------
    k_gemm_bf16_wmma<<<dim3(Mpad / 128, 64 / 64), dim3(256), 0, stream>>>(
        xb, w3b, hbuf, 64, 256);
    hipMemsetAsync(abuf, 0, (size_t)M * 64 * 4, stream);
    k_edge_agg<<<dim3(EDGE_BLOCKS), dim3(256), 0, stream>>>(hbuf, abuf, dinv, src, dst, E, 64);
    k_post_softmax<<<dim3((M + 7) / 8), dim3(256), 0, stream>>>(hbuf, abuf, dinv, b3, out, M);
}
